// ProbSparseSelfAttention_65386582114983
// MI455X (gfx1250) — matrix operand order repaired and checked by reading, not compiled, not run on hardware
//
#include <hip/hip_runtime.h>
#include <stdint.h>

// ---------- types ----------
typedef __attribute__((ext_vector_type(16))) __bf16        v16bf;
typedef __attribute__((ext_vector_type(8)))  float         v8f;
typedef __attribute__((ext_vector_type(4)))  unsigned int  v4u;
typedef __attribute__((ext_vector_type(8)))  unsigned int  v8u;
typedef __attribute__((ext_vector_type(4)))  float         v4f;

// ---------- problem constants ----------
#define B_   2
#define L_   2048
#define D_   1024
#define H_   16
#define HD_  64
#define M_   (B_ * L_)   // 4096 rows
#define KTOP 409         // int(2048 * 0.2)

// ---------- optional CDNA5 async global->LDS path ----------
#if defined(__AMDGCN__) && __has_builtin(__builtin_amdgcn_global_load_async_to_lds_b128)
#define USE_ASYNC_LDS 1
#endif

#ifdef USE_ASYNC_LDS
// builtin parameter type per compiler diagnostic: vector_size(16) int, AS1 / AS3
typedef __attribute__((__vector_size__(4 * sizeof(int)))) int a_v4i;
typedef __attribute__((address_space(1))) a_v4i* gp_v4i;
typedef __attribute__((address_space(3))) a_v4i* lp_v4i;
#endif

__device__ inline void copy16_g2l(const void* g, void* l) {
#ifdef USE_ASYNC_LDS
  void* gnc = const_cast<void*>(g);
  __builtin_amdgcn_global_load_async_to_lds_b128((gp_v4i)gnc, (lp_v4i)l, 0, 0);
#else
  *(v4u*)l = *(const v4u*)g;     // fallback: through VGPRs
#endif
}
__device__ inline void copy16_wait() {
#ifdef USE_ASYNC_LDS
#if __has_builtin(__builtin_amdgcn_s_wait_asynccnt)
  __builtin_amdgcn_s_wait_asynccnt(0);
#else
  __asm__ volatile("s_wait_asynccnt 0" ::: "memory");
#endif
#endif
}

// ---------- helpers ----------
__device__ inline unsigned short f2bf(float x) {
  unsigned int u = __float_as_uint(x);
  u += 0x7FFFu + ((u >> 16) & 1u);               // round-to-nearest-even
  return (unsigned short)(u >> 16);
}
__device__ inline float bf2f(unsigned short h) {
  return __uint_as_float(((unsigned int)h) << 16);
}

// A fragment (16xK tile, 16-bit): lanes 0-15 hold M=lane, K runs {0..7,16..23};
// lanes 16-31 hold M=lane-16, K runs {8..15,24..31}.  (ISA 7.12.2)
__device__ inline v16bf frag_a(const unsigned short* base, int stride, int ln) {
  const int hi = ln >> 4, m = ln & 15;
  const unsigned short* p = base + m * stride + hi * 8;
  v4u lo = *(const v4u*)p;            // 8 bf16
  v4u hh = *(const v4u*)(p + 16);     // 8 bf16
  v8u w;  w.lo = lo;  w.hi = hh;
  return __builtin_bit_cast(v16bf, w);
}
// B fragment from an N-major [16][K] tile (i.e. B^T layout): lane n = ln&15,
// K runs {0..7,16..23} (lanes 0-15) / {8..15,24..31} (lanes 16-31), same as A.
__device__ inline v16bf frag_b(const unsigned short* base, int stride, int ln) {
  const int hi = ln >> 4, n = ln & 15;
  const unsigned short* p = base + n * stride + hi * 8;
  v4u lo = *(const v4u*)p;
  v4u hh = *(const v4u*)(p + 16);
  v8u w;  w.lo = lo;  w.hi = hh;
  return __builtin_bit_cast(v16bf, w);
}
__device__ inline v8f wmma_bf16(v16bf a, v16bf b, v8f c) {
  return __builtin_amdgcn_wmma_f32_16x16x32_bf16(
      /*neg_a=*/false, a, /*neg_b=*/false, b,
      /*c_mod=*/(short)0, c, /*reuse_a=*/false, /*reuse_b=*/false);
}

// ---------- GEMM: Y = X @ W^T + b  (M_=4096 x N=1024, K=1024), BK=64 ----------
// MODE 0: A = f32, output = bf16 head-split [B,H,L,HD]
// MODE 1: A = bf16, output = f32 flat [M_, D_]
template <int MODE>
__global__ __launch_bounds__(256)
void gemm_bf16_kernel(const void* __restrict__ Ap, const float* __restrict__ W,
                      const float* __restrict__ bias, void* __restrict__ Outp) {
  __shared__ __align__(16) unsigned short As[32 * 64];   // 4 KB
  __shared__ __align__(16) unsigned short Bs[64 * 64];   // 8 KB
  const int tid  = threadIdx.x;
  const int ln   = tid & 31, wave = tid >> 5;
  const int wm   = wave >> 2, wn = wave & 3;     // 2 x 4 wave grid of 16x16 tiles
  const int m0   = blockIdx.x * 32;
  const int n0   = blockIdx.y * 64;

  v8f c;
  {
    const float bv = bias[n0 + wn * 16 + (ln & 15)];
#pragma unroll
    for (int i = 0; i < 8; ++i) c[i] = bv;
  }

  for (int k0 = 0; k0 < D_; k0 += 64) {
    if (MODE == 0) {
      const float* A = (const float*)Ap;
#pragma unroll
      for (int j = 0; j < 8; ++j) {
        int li = tid + j * 256, r = li >> 6, cc = li & 63;
        As[li] = f2bf(A[(size_t)(m0 + r) * D_ + k0 + cc]);
      }
      if (k0 + 64 < D_)                           // prefetch next A slab
        __builtin_prefetch(&A[(size_t)(m0 + (tid & 31)) * D_ + k0 + 64], 0, 1);
    } else {
      const unsigned short* A = (const unsigned short*)Ap;
#pragma unroll
      for (int j = 0; j < 8; ++j) {
        int li = tid + j * 256, r = li >> 6, cc = li & 63;
        As[li] = A[(size_t)(m0 + r) * D_ + k0 + cc];
      }
      if (k0 + 64 < D_)
        __builtin_prefetch(&A[(size_t)(m0 + (tid & 31)) * D_ + k0 + 64], 0, 1);
    }
#pragma unroll
    for (int j = 0; j < 16; ++j) {
      int li = tid + j * 256, r = li >> 6, cc = li & 63;
      Bs[li] = f2bf(W[(size_t)(n0 + r) * D_ + k0 + cc]);   // W[n][k] rows = B^T
    }
    if (k0 + 64 < D_)                             // prefetch next W slab
      __builtin_prefetch(&W[(size_t)(n0 + (tid >> 2)) * D_ + k0 + 64], 0, 1);
    __syncthreads();
#pragma unroll
    for (int ks = 0; ks < 2; ++ks) {
      v16bf a = frag_a(As + wm * 16 * 64 + ks * 32, 64, ln);
      v16bf b = frag_b(Bs + wn * 16 * 64 + ks * 32, 64, ln);
      c = wmma_bf16(a, b, c);
    }
    __syncthreads();
  }

  const int hi = ln >> 4, col = ln & 15;
#pragma unroll
  for (int r = 0; r < 8; ++r) {
    const int gm  = m0 + wm * 16 + r + 8 * hi;
    const int gn  = n0 + wn * 16 + col;
    const float v = c[r];
    if (MODE == 0) {
      unsigned short* O = (unsigned short*)Outp;
      const int b = gm >> 11, l = gm & (L_ - 1);
      const int h = gn >> 6, d = gn & (HD_ - 1);
      O[(((size_t)(b * H_ + h)) * L_ + l) * HD_ + d] = f2bf(v);
    } else {
      float* O = (float*)Outp;
      O[(size_t)gm * D_ + gn] = v;
    }
  }
}

// ---------- q-norm: ||q||^2 per (b,h,l) row ----------
__global__ __launch_bounds__(256)
void qnorm_kernel(const unsigned short* __restrict__ Q, float* __restrict__ norms) {
  const int row = blockIdx.x * 256 + threadIdx.x;      // < 65536
  const v4u* q  = (const v4u*)(Q + (size_t)row * HD_);
  float s = 0.f;
#pragma unroll
  for (int j = 0; j < 8; ++j) {
    v4u u = q[j];
#pragma unroll
    for (int t = 0; t < 4; ++t) {
      unsigned int w = u[t];
      float a = bf2f((unsigned short)(w & 0xFFFFu));
      float b = bf2f((unsigned short)(w >> 16));
      s += a * a + b * b;
    }
  }
  norms[row] = s;
}

// ---------- top-k threshold per (b,h) via binary search on count ----------
__global__ __launch_bounds__(256)
void topk_thresh_kernel(const float* __restrict__ norms, float* __restrict__ thr) {
  __shared__ float red[256];
  const int tid = threadIdx.x;
  const float* n = norms + (size_t)blockIdx.x * L_;
  float v[8];
#pragma unroll
  for (int j = 0; j < 8; ++j) v[j] = n[tid + j * 256];
  float mx = v[0];
#pragma unroll
  for (int j = 1; j < 8; ++j) mx = fmaxf(mx, v[j]);
  red[tid] = mx;
  __syncthreads();
  for (int s = 128; s > 0; s >>= 1) {
    if (tid < s) red[tid] = fmaxf(red[tid], red[tid + s]);
    __syncthreads();
  }
  float hiv = red[0];
  __syncthreads();
  float lo = 0.f;
  for (int it = 0; it < 25; ++it) {
    const float mid = 0.5f * (lo + hiv);
    float c = 0.f;
#pragma unroll
    for (int j = 0; j < 8; ++j) c += (v[j] > mid) ? 1.f : 0.f;
    red[tid] = c;
    __syncthreads();
    for (int s = 128; s > 0; s >>= 1) {
      if (tid < s) red[tid] += red[tid + s];
      __syncthreads();
    }
    const float tot = red[0];
    __syncthreads();
    if (tot > (float)KTOP) lo = mid; else hiv = mid;
  }
  if (tid == 0) thr[blockIdx.x] = hiv;
}

// ---------- zero non-selected query rows ----------
__global__ __launch_bounds__(256)
void mask_q_kernel(unsigned short* __restrict__ Q, const float* __restrict__ norms,
                   const float* __restrict__ thr) {
  const int row = blockIdx.x * 256 + threadIdx.x;
  const int bh  = row >> 11;
  if (!(norms[row] > thr[bh])) {
    v4u z = (v4u)(0u);
    v4u* p = (v4u*)(Q + (size_t)row * HD_);
#pragma unroll
    for (int j = 0; j < 8; ++j) p[j] = z;
  }
}

// ---------- fused flash attention (zeroed rows -> uniform softmax) ----------
__global__ __launch_bounds__(128)
void flash_attn_kernel(const unsigned short* __restrict__ Qw,
                       const unsigned short* __restrict__ Kw,
                       const unsigned short* __restrict__ Vw,
                       unsigned short* __restrict__ Ow) {
  __shared__ __align__(16) unsigned short Qs[64 * 64];   // [qrow][d]
  __shared__ __align__(16) unsigned short Ks[32 * 64];   // [kvrow][d] (= B^T for scores)
  __shared__ __align__(16) unsigned short VTs[64 * 32];  // [d][kvrow] (= B^T for PV)
  __shared__ __align__(16) float          Sf[4][16 * 32];
  __shared__ __align__(16) unsigned short Pb[4][16 * 32];
  __shared__ __align__(16) float mrow[4][16], lrow[4][16], arow[4][16];

  const int tid = threadIdx.x, ln = tid & 31, wave = tid >> 5;
  const int hi  = ln >> 4, col = ln & 15;
  const int bh  = blockIdx.y;
  const int q0  = blockIdx.x * 64;
  const size_t base = (size_t)bh * L_ * HD_;
  const float scale = 0.125f;                            // HD^-0.5

  // stage 64x64 Q tile: byte-contiguous -> async global->LDS (512 x 16B chunks)
  {
    const char* src = (const char*)(Qw + base + (size_t)q0 * HD_);
    char* dst = (char*)Qs;
#pragma unroll
    for (int j = 0; j < 4; ++j) {
      const int idx = tid + j * 128;
      copy16_g2l(src + idx * 16, dst + idx * 16);
    }
  }
  if (ln < 16) { mrow[wave][ln] = -3.0e38f; lrow[wave][ln] = 0.f; }
  copy16_wait();
  __syncthreads();

  v16bf qfrag[2];
#pragma unroll
  for (int ks = 0; ks < 2; ++ks)
    qfrag[ks] = frag_a(Qs + wave * 16 * 64 + ks * 32, 64, ln);

  v8f o[4];
#pragma unroll
  for (int ch = 0; ch < 4; ++ch)
#pragma unroll
    for (int i = 0; i < 8; ++i) o[ch][i] = 0.f;

  for (int kv0 = 0; kv0 < L_; kv0 += 32) {
    __syncthreads();
    {                                                    // K tile: async (256 x 16B)
      const char* src = (const char*)(Kw + base + (size_t)kv0 * HD_);
      char* dst = (char*)Ks;
#pragma unroll
      for (int j = 0; j < 2; ++j) {
        const int idx = tid + j * 128;
        copy16_g2l(src + idx * 16, dst + idx * 16);
      }
    }
#pragma unroll
    for (int j = 0; j < 16; ++j) {                       // V tile transposed (manual)
      int li = tid + j * 128, kv = li >> 6, d = li & 63;
      VTs[d * 32 + kv] = Vw[base + (size_t)(kv0 + kv) * HD_ + d];
    }
    copy16_wait();
    __syncthreads();

    // S = Q(16x64) @ K^T(64x32) : two N-halves, K-dim 64 -> 2 wmma steps each
    v8f s0, s1;
#pragma unroll
    for (int i = 0; i < 8; ++i) { s0[i] = 0.f; s1[i] = 0.f; }
#pragma unroll
    for (int ks = 0; ks < 2; ++ks) {
      v16bf b0 = frag_b(Ks + 0 * 64 + ks * 32, 64, ln);
      v16bf b1 = frag_b(Ks + 16 * 64 + ks * 32, 64, ln);
      s0 = wmma_bf16(qfrag[ks], b0, s0);
      s1 = wmma_bf16(qfrag[ks], b1, s1);
    }

    float* st = Sf[wave];
#pragma unroll
    for (int r = 0; r < 8; ++r) {
      const int m = r + 8 * hi;
      st[m * 32 + col]      = s0[r] * scale;
      st[m * 32 + 16 + col] = s1[r] * scale;
    }
    __asm__ volatile("s_wait_dscnt 0" ::: "memory");     // intra-wave LDS ordering

    if (ln < 16) {                                       // per-row online softmax
      const int m = ln;
      const float mo = mrow[wave][m];
      float mx = mo;
      float sv[32];
#pragma unroll
      for (int cix = 0; cix < 32; ++cix) { sv[cix] = st[m * 32 + cix]; mx = fmaxf(mx, sv[cix]); }
      const float al = __expf(mo - mx);
      float sum = 0.f;
      unsigned short* pb = Pb[wave] + m * 32;
#pragma unroll
      for (int cix = 0; cix < 32; ++cix) {
        const float e = __expf(sv[cix] - mx);
        sum += e;
        pb[cix] = f2bf(e);
      }
      mrow[wave][m] = mx;
      lrow[wave][m] = lrow[wave][m] * al + sum;
      arow[wave][m] = al;
    }
    __asm__ volatile("s_wait_dscnt 0" ::: "memory");

    const v4f a0 = *(const v4f*)&arow[wave][8 * hi];
    const v4f a1 = *(const v4f*)&arow[wave][8 * hi + 4];
    float al8[8] = {a0.x, a0.y, a0.z, a0.w, a1.x, a1.y, a1.z, a1.w};
#pragma unroll
    for (int ch = 0; ch < 4; ++ch)
#pragma unroll
      for (int r = 0; r < 8; ++r) o[ch][r] *= al8[r];

    // O += P(16x32) @ V(32x64) : 4 N-chunks of 16
    const v16bf pa = frag_a(Pb[wave], 32, ln);
#pragma unroll
    for (int ch = 0; ch < 4; ++ch) {
      v16bf bv = frag_b(VTs + ch * 16 * 32, 32, ln);
      o[ch] = wmma_bf16(pa, bv, o[ch]);
    }
  }

  const v4f l0 = *(const v4f*)&lrow[wave][8 * hi];
  const v4f l1 = *(const v4f*)&lrow[wave][8 * hi + 4];
  const float li8[8] = {l0.x, l0.y, l0.z, l0.w, l1.x, l1.y, l1.z, l1.w};
  const int b = bh >> 4, h = bh & 15;
#pragma unroll
  for (int ch = 0; ch < 4; ++ch)
#pragma unroll
    for (int r = 0; r < 8; ++r) {
      const int gl = q0 + wave * 16 + r + 8 * hi;
      const int d  = ch * 16 + col;
      const float v = o[ch][r] / li8[r];
      Ow[((size_t)b * L_ + gl) * D_ + h * HD_ + d] = f2bf(v);   // merged heads [B,L,D]
    }
}

// ---------- launcher ----------
extern "C" void kernel_launch(void* const* d_in, const int* in_sizes, int n_in,
                              void* d_out, int out_size, void* d_ws, size_t ws_size,
                              hipStream_t stream) {
  const float* x  = (const float*)d_in[0];
  const float* Wq = (const float*)d_in[1];
  const float* bq = (const float*)d_in[2];
  const float* Wk = (const float*)d_in[3];
  const float* bk = (const float*)d_in[4];
  const float* Wv = (const float*)d_in[5];
  const float* bv = (const float*)d_in[6];
  const float* Wo = (const float*)d_in[7];
  const float* bo = (const float*)d_in[8];

  char* ws = (char*)d_ws;                                // 33.8 MB used
  unsigned short* Qw = (unsigned short*)(ws);            // 8 MiB [B,H,L,HD] bf16
  unsigned short* Kw = (unsigned short*)(ws + (8u << 20));
  unsigned short* Vw = (unsigned short*)(ws + (16u << 20));
  unsigned short* Aw = (unsigned short*)(ws + (24u << 20)); // attn out [B,L,D] bf16
  float* norms = (float*)(ws + (32u << 20));             // 65536 f32
  float* thr   = (float*)(ws + (32u << 20) + 65536 * sizeof(float)); // 32 f32

  const dim3 g(M_ / 32, D_ / 64), blk(256);
  gemm_bf16_kernel<0><<<g, blk, 0, stream>>>(x, Wq, bq, Qw);
  gemm_bf16_kernel<0><<<g, blk, 0, stream>>>(x, Wk, bk, Kw);
  gemm_bf16_kernel<0><<<g, blk, 0, stream>>>(x, Wv, bv, Vw);
  qnorm_kernel<<<256, 256, 0, stream>>>(Qw, norms);
  topk_thresh_kernel<<<B_ * H_, 256, 0, stream>>>(norms, thr);
  mask_q_kernel<<<256, 256, 0, stream>>>(Qw, norms, thr);
  flash_attn_kernel<<<dim3(L_ / 64, B_ * H_), 128, 0, stream>>>(Qw, Kw, Vw, Aw);
  gemm_bf16_kernel<1><<<g, blk, 0, stream>>>(Aw, Wo, bo, d_out);
}
